// LowMixer_67499706024700
// MI455X (gfx1250) — matrix operand order repaired and checked by reading, not compiled, not run on hardware
//
#include <hip/hip_runtime.h>

// ---------------------------------------------------------------------------
// MI455X (gfx1250, wave32) implementation of the dual-attention "LowMixer".
// Memory-bound problem (~0.3 GB traffic vs ~24 GFLOP): everything runs in
// f16 WMMA (v_wmma_f32_16x16x32_f16) with fused reshape/upsample epilogue.
// CDNA5 async global->LDS copies (ASYNCcnt) stage the contiguous K-tile and
// GEMM-A-tile; builtin signature per clang: (int4 AS1*, int4 AS3*, imm, imm).
// ---------------------------------------------------------------------------

typedef __attribute__((ext_vector_type(16))) _Float16 v16h;
typedef __attribute__((ext_vector_type(8)))  _Float16 v8h;
typedef __attribute__((ext_vector_type(8)))  float    v8f;
typedef __attribute__((ext_vector_type(4)))  int      v4i;
typedef _Float16 h16;

union AF { v16h v; v8h h[2]; };

constexpr int Bb   = 8;
constexpr int Cc   = 256;
constexpr int Himg = 64;
constexpr int Wimg = 64;
constexpr int NH   = 8;
constexpr int HD   = 32;
constexpr int Ntok = 1024;           // 32x32 pooled tokens per batch
constexpr int TOK  = Bb * Ntok;      // 8192 token rows total

#if defined(__gfx1250__) && __has_builtin(__builtin_amdgcn_global_load_async_to_lds_b128)
#define HAVE_ASYNC_LDS 1
#else
#define HAVE_ASYNC_LDS 0
#endif

#if HAVE_ASYNC_LDS
// param0: global src (int4 addrspace(1)*), param1: LDS dst (int4 addrspace(3)*)
#define GAS(p) ((__attribute__((address_space(1))) v4i*)(p))
#define LAS(p) ((__attribute__((address_space(3))) v4i*)(p))
__device__ __forceinline__ void async_wait0() {
#if __has_builtin(__builtin_amdgcn_s_wait_asynccnt)
  __builtin_amdgcn_s_wait_asynccnt(0);
#else
  asm volatile("s_wait_asynccnt 0x0" ::: "memory");
#endif
}
#endif

// ---------------------------------------------------------------------------
// Kernel 1: 2x2 avg pool + f32->f16, (B,C,H,W) -> token-major (which,B,N,C).
// LDS transpose so global reads are row-runs and writes are 128-bit, C-contig.
// ---------------------------------------------------------------------------
__global__ void __launch_bounds__(256)
pool_kernel(const float* __restrict__ x1, const float* __restrict__ x2,
            h16* __restrict__ xa) {
  __shared__ __align__(16) h16 ps[32][264];   // [col][c], 528B row = 33*16B
  const int b = blockIdx.x >> 5;
  const int r = blockIdx.x & 31;              // pooled row
  const int which = blockIdx.y;
  const float* xs = which ? x2 : x1;
  const int c = threadIdx.x;                  // channel
  const float* row0 = xs + ((size_t)(b * Cc + c) * Himg + 2 * r) * Wimg;
#pragma unroll
  for (int col = 0; col < 32; ++col) {
    float2 a = *(const float2*)(row0 + 2 * col);
    float2 d = *(const float2*)(row0 + Wimg + 2 * col);
    ps[col][c] = (h16)((a.x + a.y + d.x + d.y) * 0.25f);
  }
  __syncthreads();
  h16* dst = xa + (size_t)which * TOK * Cc + ((size_t)b * Ntok + r * 32) * Cc;
  const int lane = threadIdx.x & 31;
  const int cg = threadIdx.x >> 5;
#pragma unroll
  for (int p = 0; p < 4; ++p) {
    int col = p * 8 + cg;
    v8h v = *(const v8h*)&ps[col][lane * 8];
    *(v8h*)(dst + col * Cc + lane * 8) = v;   // coalesced b128 token-major
  }
}

// ---------------------------------------------------------------------------
// Kernel 2: WMMA GEMM, M=8192 tokens, K=256, Nout=768 = [K(256)|V(256)|Qraw(256)].
// Block tile 128x64, 8 waves each computing 32x32 (4 WMMA accs, K-chunks of 32).
// Outputs scattered into (b,n,h,d) f16 so attention fragments read contiguously.
// A-tile staging uses async global->LDS (contiguous rows; LDS pad preserved by
// issuing two b128 asyncs with imm offsets 0/16, which apply to both addrs).
// ---------------------------------------------------------------------------
__global__ void __launch_bounds__(256)
qkv_gemm_kernel(const h16* __restrict__ xa, const float* __restrict__ Wqkv,
                const float* __restrict__ Wq, h16* __restrict__ kh,
                h16* __restrict__ vh, h16* __restrict__ qraw) {
  __shared__ __align__(16) h16 As[128 * 40];  // [row][c], 80B stride
  __shared__ __align__(16) h16 Bs[64 * 40];   // [j][c] transposed weights
  const int Mt = blockIdx.x * 128;
  const int Jt = blockIdx.y * 64;
  const int which = blockIdx.z;
  const size_t xoff = (size_t)which * TOK * Cc;
  const h16* A = xa + xoff;
  const int t = threadIdx.x;
  const int lane = t & 31, wid = t >> 5;
  const int msub = (wid >> 1) * 32, jsub = (wid & 1) * 32;
  const int hi = lane >> 4, l16 = lane & 15;
  v8f acc[2][2] = {};

  for (int kc = 0; kc < 256; kc += 32) {
    __syncthreads();
    {  // stage A tile: 128 rows x 32 halves
      int row = t >> 1, seg = (t & 1) * 16;
      const h16* src = A + (size_t)(Mt + row) * Cc + kc + seg;
      h16* dst = &As[row * 40 + seg];
#if HAVE_ASYNC_LDS
      __builtin_amdgcn_global_load_async_to_lds_b128(GAS(src), LAS(dst), 0, 0);
      __builtin_amdgcn_global_load_async_to_lds_b128(GAS(src), LAS(dst), 16, 0);
#else
      v8h a0 = ((const v8h*)src)[0], a1 = ((const v8h*)src)[1];
      *(v8h*)dst = a0;
      *(v8h*)(dst + 8) = a1;
#endif
    }
    {  // stage B transposed: Bs[j][c] (f32 -> f16)
      int cc2 = t >> 3;                // K index within chunk
      int jg  = (t & 7) * 8;           // 8 consecutive output cols
      int j = Jt + jg;
      const float* src = (j < 512) ? (Wqkv + (size_t)(kc + cc2) * 512 + j)
                                   : (Wq + (size_t)(kc + cc2) * 256 + (j - 512));
      float4 w0 = *(const float4*)(src);
      float4 w1 = *(const float4*)(src + 4);
      h16* d = &Bs[jg * 40 + cc2];
      d[0 * 40] = (h16)w0.x; d[1 * 40] = (h16)w0.y;
      d[2 * 40] = (h16)w0.z; d[3 * 40] = (h16)w0.w;
      d[4 * 40] = (h16)w1.x; d[5 * 40] = (h16)w1.y;
      d[6 * 40] = (h16)w1.z; d[7 * 40] = (h16)w1.w;
    }
#if HAVE_ASYNC_LDS
    async_wait0();
#endif
    __syncthreads();
    AF a[2], bf[2];
#pragma unroll
    for (int i = 0; i < 2; ++i) {
      int row = msub + i * 16 + l16;   // A: lane = M row, K split by lane-half
      a[i].h[0] = *(const v8h*)&As[row * 40 + hi * 16];
      a[i].h[1] = *(const v8h*)&As[row * 40 + hi * 16 + 8];
      int jl = jsub + i * 16 + l16;    // B: lane = N col, 16 contiguous K halves
      bf[i].h[0] = *(const v8h*)&Bs[jl * 40 + hi * 16];
      bf[i].h[1] = *(const v8h*)&Bs[jl * 40 + hi * 16 + 8];
    }
#pragma unroll
    for (int i = 0; i < 2; ++i)
#pragma unroll
      for (int j2 = 0; j2 < 2; ++j2)
        acc[i][j2] = __builtin_amdgcn_wmma_f32_16x16x32_f16(
            false, a[i].v, false, bf[j2].v, (short)0, acc[i][j2], false, false);
  }
  // epilogue: split cols into k / v / q-raw, f16, (b,n,h,d) layout
  h16* kd = kh + xoff;
  h16* vd = vh + xoff;
  h16* qd = qraw + xoff;
#pragma unroll
  for (int i = 0; i < 2; ++i)
#pragma unroll
    for (int j2 = 0; j2 < 2; ++j2) {
      int colg = Jt + jsub + j2 * 16 + l16;
#pragma unroll
      for (int j = 0; j < 8; ++j) {
        int bn = Mt + msub + i * 16 + j + hi * 8;
        h16 val = (h16)acc[i][j2][j];
        if (colg < 256)      kd[(size_t)bn * 256 + colg] = val;
        else if (colg < 512) vd[(size_t)bn * 256 + (colg - 256)] = val;
        else                 qd[(size_t)bn * 256 + (colg - 512)] = val;
      }
    }
}

// ---------------------------------------------------------------------------
// Kernel 3: gate q = (1 - sigmoid(q1*q2)) * scale * log2(e)  (scale folded so
// attention softmax runs in base-2 with plain exp2).
// ---------------------------------------------------------------------------
__global__ void __launch_bounds__(256)
gate_kernel(const h16* __restrict__ q1r, const h16* __restrict__ q2r,
            h16* __restrict__ qg) {
  const float SC = 0.17677669529663687f * 1.4426950408889634f;  // hd^-0.5 * log2e
  size_t i = ((size_t)blockIdx.x * 256 + threadIdx.x) * 8;
  v8h a = *(const v8h*)(q1r + i);
  v8h b = *(const v8h*)(q2r + i);
  v8h o;
#pragma unroll
  for (int j = 0; j < 8; ++j) {
    float p = (float)a[j] * (float)b[j];
    float g = 1.0f - 1.0f / (1.0f + __expf(-p));
    o[j] = (h16)(g * SC);
  }
  *(v8h*)(qg + i) = o;
}

// ---------------------------------------------------------------------------
// Kernel 4: flash attention per (pass,b,h). Wave owns 16 query rows; K/V^T
// staged in LDS per 128-key chunk (K via async global->LDS, V via manual
// transpose); online base-2 softmax with half-wave shfl reductions; p relaid
// out C->A via per-wave LDS tile; epilogue fuses the torch-faithful reshape
// (c=n>>2, r=(n&3)*8+h, col=d) + 2x2 nearest upsample.
// ---------------------------------------------------------------------------
__global__ void __launch_bounds__(256)
attn_kernel(const h16* __restrict__ qg, const h16* __restrict__ kh,
            const h16* __restrict__ vh, float* __restrict__ out) {
  __shared__ __align__(16) h16 kS[128 * 40];      // [key][d], 80B stride
  __shared__ __align__(16) h16 vT[32 * 136];      // [d][key], 272B stride
  __shared__ __align__(16) h16 stg[8][16][32];    // per-wave p relayout tile
  const int pass = blockIdx.y;
  const int nblk = blockIdx.x & 7;
  const int bh = blockIdx.x >> 3;
  const int h = bh & 7;
  const int b = bh >> 3;
  const int t = threadIdx.x;
  const int lane = t & 31, wid = t >> 5;
  const int hi = lane >> 4, l16 = lane & 15;
  const h16* kbase = kh + (size_t)pass * TOK * Cc;
  const h16* vbase = vh + (size_t)pass * TOK * Cc;
  const int nb = nblk * 128 + wid * 16;           // first query row of this wave

  AF aq;  // q A-fragment, loaded once (scale/log2e pre-folded)
  {
    const h16* qrow = qg + ((size_t)(b * Ntok + nb + l16) * NH + h) * HD;
    aq.h[0] = *(const v8h*)(qrow + hi * 16);
    aq.h[1] = *(const v8h*)(qrow + hi * 16 + 8);
  }
  float m[8], rs[8];
  v8f acc0 = {}, acc1 = {};
#pragma unroll
  for (int j = 0; j < 8; ++j) { m[j] = -1e30f; rs[j] = 0.0f; }

  for (int m0 = 0; m0 < Ntok; m0 += 128) {
    __syncthreads();
    {  // stage K chunk (row-major: B-frag for q·k^T reads contiguous d runs)
      int key = t >> 1, seg = (t & 1) * 16;
      const h16* src = kbase + ((size_t)(b * Ntok + m0 + key) * NH + h) * HD + seg;
      h16* dst = &kS[key * 40 + seg];
#if HAVE_ASYNC_LDS
      __builtin_amdgcn_global_load_async_to_lds_b128(GAS(src), LAS(dst), 0, 0);
      __builtin_amdgcn_global_load_async_to_lds_b128(GAS(src), LAS(dst), 16, 0);
#else
      v8h k0 = ((const v8h*)src)[0], k1 = ((const v8h*)src)[1];
      *(v8h*)dst = k0;
      *(v8h*)(dst + 8) = k1;
#endif
    }
    {  // stage V transposed (B-frag for p·v reads contiguous key runs)
      int key = t & 127, dg = (t >> 7) * 16;
      const v8h* src = (const v8h*)(vbase +
          ((size_t)(b * Ntok + m0 + key) * NH + h) * HD + dg);
      v8h v0 = src[0], v1 = src[1];
#pragma unroll
      for (int i = 0; i < 8; ++i) {
        vT[(dg + i) * 136 + key] = v0[i];
        vT[(dg + 8 + i) * 136 + key] = v1[i];
      }
    }
#if HAVE_ASYNC_LDS
    async_wait0();
#endif
    __syncthreads();

    for (int mo = 0; mo < 128; mo += 32) {
      AF bk0, bk1;
      bk0.h[0] = *(const v8h*)&kS[(mo + l16) * 40 + hi * 16];
      bk0.h[1] = *(const v8h*)&kS[(mo + l16) * 40 + hi * 16 + 8];
      bk1.h[0] = *(const v8h*)&kS[(mo + 16 + l16) * 40 + hi * 16];
      bk1.h[1] = *(const v8h*)&kS[(mo + 16 + l16) * 40 + hi * 16 + 8];
      v8f zero = {};
      v8f s0 = __builtin_amdgcn_wmma_f32_16x16x32_f16(
          false, aq.v, false, bk0.v, (short)0, zero, false, false);
      v8f s1 = __builtin_amdgcn_wmma_f32_16x16x32_f16(
          false, aq.v, false, bk1.v, (short)0, zero, false, false);
      // online softmax over this 32-key step (base-2 domain)
#pragma unroll
      for (int j = 0; j < 8; ++j) {
        float tm = fmaxf(s0[j], s1[j]);
#pragma unroll
        for (int msk = 1; msk < 16; msk <<= 1)
          tm = fmaxf(tm, __shfl_xor(tm, msk, 32));
        float nm = fmaxf(m[j], tm);
        float corr = exp2f(m[j] - nm);
        m[j] = nm;
        float p0 = exp2f(s0[j] - nm);
        float p1 = exp2f(s1[j] - nm);
        float ts = p0 + p1;
#pragma unroll
        for (int msk = 1; msk < 16; msk <<= 1)
          ts += __shfl_xor(ts, msk, 32);
        rs[j] = rs[j] * corr + ts;
        acc0[j] *= corr;
        acc1[j] *= corr;
        int r = j + hi * 8;                 // C-layout row this lane holds
        stg[wid][r][l16] = (h16)p0;         // relayout p: C-layout -> LDS
        stg[wid][r][16 + l16] = (h16)p1;
      }
      // reload p as A-fragment (wave-private tile; in-wave DS ordering)
      AF ap, bv0, bv1;
      ap.h[0] = *(const v8h*)&stg[wid][l16][hi * 16];
      ap.h[1] = *(const v8h*)&stg[wid][l16][hi * 16 + 8];
      bv0.h[0] = *(const v8h*)&vT[l16 * 136 + mo + hi * 16];
      bv0.h[1] = *(const v8h*)&vT[l16 * 136 + mo + hi * 16 + 8];
      bv1.h[0] = *(const v8h*)&vT[(16 + l16) * 136 + mo + hi * 16];
      bv1.h[1] = *(const v8h*)&vT[(16 + l16) * 136 + mo + hi * 16 + 8];
      acc0 = __builtin_amdgcn_wmma_f32_16x16x32_f16(
          false, ap.v, false, bv0.v, (short)0, acc0, false, false);
      acc1 = __builtin_amdgcn_wmma_f32_16x16x32_f16(
          false, ap.v, false, bv1.v, (short)0, acc1, false, false);
    }
  }

  // epilogue: normalize + fused reshape(B,C,N) + 2x nearest upsample
  float* ob = out + (size_t)pass * Bb * Cc * Himg * Wimg;
#pragma unroll
  for (int j = 0; j < 8; ++j) {
    int n = nb + j + hi * 8;
    float inv = 1.0f / rs[j];
    int c = n >> 2;
    int rr = (n & 3) * 8 + h;
    float* o0 = ob + ((size_t)(b * Cc + c) * Himg + 2 * rr) * Wimg;
    float v0 = acc0[j] * inv, v1 = acc1[j] * inv;
    int d0 = l16, d1 = 16 + l16;
    float2 w0 = {v0, v0}, w1 = {v1, v1};
    *(float2*)(o0 + 2 * d0) = w0;
    *(float2*)(o0 + Wimg + 2 * d0) = w0;
    *(float2*)(o0 + 2 * d1) = w1;
    *(float2*)(o0 + Wimg + 2 * d1) = w1;
  }
}

// ---------------------------------------------------------------------------
extern "C" void kernel_launch(void* const* d_in, const int* in_sizes, int n_in,
                              void* d_out, int out_size, void* d_ws, size_t ws_size,
                              hipStream_t stream) {
  const float* x1   = (const float*)d_in[0];
  const float* x2   = (const float*)d_in[1];
  const float* Wqkv = (const float*)d_in[2];
  const float* Wq   = (const float*)d_in[3];
  float* out = (float*)d_out;

  // workspace layout (36 MB total): all f16
  char* ws = (char*)d_ws;
  h16* xa   = (h16*)(ws);                               // 2 x 8192 x 256
  h16* kh   = (h16*)(ws + (size_t)8 * 1024 * 1024);     // 2 x 8192 x 256
  h16* vh   = (h16*)(ws + (size_t)16 * 1024 * 1024);    // 2 x 8192 x 256
  h16* qraw = (h16*)(ws + (size_t)24 * 1024 * 1024);    // 2 x 8192 x 256
  h16* qg   = (h16*)(ws + (size_t)32 * 1024 * 1024);    // 8192 x 256 (shared)

  pool_kernel<<<dim3(Bb * 32, 2), 256, 0, stream>>>(x1, x2, xa);
  qkv_gemm_kernel<<<dim3(64, 12, 2), 256, 0, stream>>>(xa, Wqkv, Wq, kh, vh, qraw);
  gate_kernel<<<dim3(1024), 256, 0, stream>>>(qraw, qraw + (size_t)TOK * Cc, qg);
  attn_kernel<<<dim3(512, 2), 256, 0, stream>>>(qg, kh, vh, out);
}
